// FlashMultiheadAttention_52441550684580
// MI455X (gfx1250) — matrix operand order repaired and checked by reading, not compiled, not run on hardware
//
#include <hip/hip_runtime.h>

typedef __attribute__((ext_vector_type(16))) __bf16 v16bf;
typedef __attribute__((ext_vector_type(8)))  float  v8f;

#define HEADS    16
#define HEAD_DIM 64
#define SEQ      2048
#define BATCH    4
#define CDIM     1024
#define ROWS     (BATCH * SEQ)        // 8192
#define ATT_SCALE 0.125f              // 1/sqrt(64)

__device__ __forceinline__ unsigned short f32_to_bf16(float f) {
    union { float f; unsigned u; } v; v.f = f;
    unsigned u = v.u;
    unsigned r = u + 0x7FFFu + ((u >> 16) & 1u);   // round-to-nearest-even
    return (unsigned short)(r >> 16);
}

union FragAB {                        // 16 bf16 = one WMMA A or B operand
    v16bf bf;
    unsigned short us[16];
    uint4 q[2];
};

// ---------------------------------------------------------------- prep ----
__global__ void cvt_f32_bf16(const float* __restrict__ src,
                             unsigned short* __restrict__ dst, int count) {
    int i = blockIdx.x * blockDim.x + threadIdx.x;
    if (i < count) dst[i] = f32_to_bf16(src[i]);
}

// src [K][F] row-major f32  ->  dst [F][K] bf16
__global__ void transpose_cvt(const float* __restrict__ src,
                              unsigned short* __restrict__ dst, int K, int F) {
    int i = blockIdx.x * blockDim.x + threadIdx.x;
    if (i < K * F) {
        int f = i / K, k = i - f * K;
        dst[i] = f32_to_bf16(src[(size_t)k * F + f]);
    }
}

// ------------------------------------------------------------ QKV GEMM ----
// xbf [8192][1024] bf16, wT [3072][1024] bf16 (pre-transposed)
// -> q,k [B,H,N,D] bf16 ; vt [B,H,D,N] bf16
// WG = 4 waves; wave tile = 32(M) x 64(N); WG tile = 128 x 64
__global__ void qkv_gemm(const unsigned short* __restrict__ xbf,
                         const unsigned short* __restrict__ wT,
                         unsigned short* __restrict__ qbuf,
                         unsigned short* __restrict__ kbuf,
                         unsigned short* __restrict__ vtbuf) {
    const int wv = threadIdx.x >> 5;
    const int lane = threadIdx.x & 31;
    const int g = lane >> 4;           // half-wave group
    const int lr = lane & 15;
    const int m0 = blockIdx.x * 128 + wv * 32;
    const int n0 = blockIdx.y * 64;

    v8f acc[2][4] = {};
    const unsigned short* arow0 = xbf + (size_t)(m0 + lr) * CDIM;
    const unsigned short* arow1 = arow0 + (size_t)16 * CDIM;
    for (int k0 = 0; k0 < CDIM; k0 += 32) {
        __builtin_prefetch(arow0 + k0 + 128, 0, 1);   // global_prefetch_b8
        __builtin_prefetch(arow1 + k0 + 128, 0, 1);
        FragAB a0, a1;
        a0.q[0] = *reinterpret_cast<const uint4*>(arow0 + k0 + g * 8);
        a0.q[1] = *reinterpret_cast<const uint4*>(arow0 + k0 + 16 + g * 8);
        a1.q[0] = *reinterpret_cast<const uint4*>(arow1 + k0 + g * 8);
        a1.q[1] = *reinterpret_cast<const uint4*>(arow1 + k0 + 16 + g * 8);
#pragma unroll
        for (int nb = 0; nb < 4; ++nb) {
            FragAB b;
            const unsigned short* bcol =
                wT + (size_t)(n0 + nb * 16 + lr) * CDIM + k0 + g * 8;
            b.q[0] = *reinterpret_cast<const uint4*>(bcol);
            b.q[1] = *reinterpret_cast<const uint4*>(bcol + 16);
            acc[0][nb] = __builtin_amdgcn_wmma_f32_16x16x32_bf16(
                false, a0.bf, false, b.bf, (short)0, acc[0][nb], false, false);
            acc[1][nb] = __builtin_amdgcn_wmma_f32_16x16x32_bf16(
                false, a1.bf, false, b.bf, (short)0, acc[1][nb], false, false);
        }
    }
    // scatter epilogue into Q, K, V^T
#pragma unroll
    for (int r = 0; r < 2; ++r) {
#pragma unroll
        for (int nb = 0; nb < 4; ++nb) {
#pragma unroll
            for (int j = 0; j < 8; ++j) {
                int m = m0 + r * 16 + j + 8 * g;   // C layout: VGPR j -> row j / j+8
                int f = n0 + nb * 16 + lr;         // column = lane % 16
                unsigned short v = f32_to_bf16(acc[r][nb][j]);
                int s   = f >> 10;                 // 0=Q 1=K 2=V
                int rem = f & 1023;
                int h = rem >> 6, d = rem & 63;
                int b = m >> 11, n = m & 2047;
                if (s == 0)
                    qbuf[((size_t)(b * HEADS + h) * SEQ + n) * HEAD_DIM + d] = v;
                else if (s == 1)
                    kbuf[((size_t)(b * HEADS + h) * SEQ + n) * HEAD_DIM + d] = v;
                else
                    vtbuf[((size_t)(b * HEADS + h) * HEAD_DIM + d) * SEQ + n] = v;
            }
        }
    }
}

// ------------------------------------------------------- flash attention ----
// one WG per (b,h,128-query block); 4 waves, 32 queries each
__global__ void flash_attn(const unsigned short* __restrict__ qbuf,
                           const unsigned short* __restrict__ kbuf,
                           const unsigned short* __restrict__ vtbuf,
                           unsigned short* __restrict__ abuf) {
    __shared__ __align__(16) unsigned short pbuf[4][32][32];
    const int wv = threadIdx.x >> 5;
    const int lane = threadIdx.x & 31;
    const int g = lane >> 4;
    const int lr = lane & 15;
    const int bh = blockIdx.x >> 4;          // b*H + h   (16 query blocks each)
    const int qb = blockIdx.x & 15;
    const int b  = bh >> 4;
    const int h  = bh & 15;
    const int m0 = qb * 128 + wv * 32;       // first query row of this wave

    // Q fragments: 2 row-blocks x 2 K-chunks of D=64 (A layout, contiguous runs)
    FragAB qf[2][2];
#pragma unroll
    for (int r = 0; r < 2; ++r) {
        const unsigned short* qrow =
            qbuf + ((size_t)bh * SEQ + (m0 + r * 16 + lr)) * HEAD_DIM;
#pragma unroll
        for (int kc = 0; kc < 2; ++kc) {
            qf[r][kc].q[0] = *reinterpret_cast<const uint4*>(qrow + kc * 32 + g * 8);
            qf[r][kc].q[1] = *reinterpret_cast<const uint4*>(qrow + kc * 32 + 16 + g * 8);
        }
    }

    v8f o[2][4] = {};
    float mrun[2][8], lrun[2][8];
#pragma unroll
    for (int r = 0; r < 2; ++r)
#pragma unroll
        for (int j = 0; j < 8; ++j) { mrun[r][j] = -1e30f; lrun[r][j] = 0.0f; }

    for (int j0 = 0; j0 < SEQ; j0 += 32) {
        // S(32x32) = Q(32x64) . K^T -- K tile shared by both row-blocks
        v8f s[2][2] = {};
#pragma unroll
        for (int kc = 0; kc < 2; ++kc) {
            FragAB kb0, kb1;
            const unsigned short* kr0 =
                kbuf + ((size_t)bh * SEQ + (j0 + lr)) * HEAD_DIM + kc * 32 + g * 8;
            const unsigned short* kr1 =
                kbuf + ((size_t)bh * SEQ + (j0 + 16 + lr)) * HEAD_DIM + kc * 32 + g * 8;
            kb0.q[0] = *reinterpret_cast<const uint4*>(kr0);
            kb0.q[1] = *reinterpret_cast<const uint4*>(kr0 + 16);
            kb1.q[0] = *reinterpret_cast<const uint4*>(kr1);
            kb1.q[1] = *reinterpret_cast<const uint4*>(kr1 + 16);
#pragma unroll
            for (int r = 0; r < 2; ++r) {
                s[r][0] = __builtin_amdgcn_wmma_f32_16x16x32_bf16(
                    false, qf[r][kc].bf, false, kb0.bf, (short)0, s[r][0], false, false);
                s[r][1] = __builtin_amdgcn_wmma_f32_16x16x32_bf16(
                    false, qf[r][kc].bf, false, kb1.bf, (short)0, s[r][1], false, false);
            }
        }
        // online softmax (row lives on 16 lanes of one half-wave group)
#pragma unroll
        for (int r = 0; r < 2; ++r) {
            float p0[8], p1[8];
#pragma unroll
            for (int j = 0; j < 8; ++j) {
                float a0 = s[r][0][j] * ATT_SCALE;
                float a1 = s[r][1][j] * ATT_SCALE;
                float mx = fmaxf(a0, a1);
#pragma unroll
                for (int off = 1; off < 16; off <<= 1)
                    mx = fmaxf(mx, __shfl_xor(mx, off, 32));
                float mnew = fmaxf(mrun[r][j], mx);
                float e0 = __expf(a0 - mnew);
                float e1 = __expf(a1 - mnew);
                float rs = e0 + e1;
#pragma unroll
                for (int off = 1; off < 16; off <<= 1)
                    rs += __shfl_xor(rs, off, 32);
                float fac = __expf(mrun[r][j] - mnew);
                lrun[r][j] = lrun[r][j] * fac + rs;
                mrun[r][j] = mnew;
                p0[j] = e0; p1[j] = e1;
#pragma unroll
                for (int db = 0; db < 4; ++db) o[r][db][j] *= fac;
            }
            // reshape P: C layout -> A layout via per-wave LDS tile
#pragma unroll
            for (int j = 0; j < 8; ++j) {
                int row = r * 16 + j + 8 * g;
                pbuf[wv][row][lr]      = f32_to_bf16(p0[j]);
                pbuf[wv][row][lr + 16] = f32_to_bf16(p1[j]);
            }
        }
        __syncthreads();
        FragAB pf[2];
#pragma unroll
        for (int r = 0; r < 2; ++r) {
            pf[r].q[0] = *reinterpret_cast<const uint4*>(&pbuf[wv][r * 16 + lr][g * 8]);
            pf[r].q[1] = *reinterpret_cast<const uint4*>(&pbuf[wv][r * 16 + lr][16 + g * 8]);
        }
        // O(32x64) += P(32x32) . V(32x64); V stored transposed [D][N]
#pragma unroll
        for (int db = 0; db < 4; ++db) {
            FragAB vf;
            const unsigned short* vr =
                vtbuf + ((size_t)bh * HEAD_DIM + db * 16 + lr) * SEQ + j0 + g * 8;
            vf.q[0] = *reinterpret_cast<const uint4*>(vr);
            vf.q[1] = *reinterpret_cast<const uint4*>(vr + 16);
#pragma unroll
            for (int r = 0; r < 2; ++r)
                o[r][db] = __builtin_amdgcn_wmma_f32_16x16x32_bf16(
                    false, pf[r].bf, false, vf.bf, (short)0, o[r][db], false, false);
        }
        __syncthreads();
    }
    // normalize and store to attn buffer [B*N][C] (= [B,N,H,D]) bf16
#pragma unroll
    for (int r = 0; r < 2; ++r) {
#pragma unroll
        for (int j = 0; j < 8; ++j) {
            float rinv = 1.0f / lrun[r][j];
            int n = m0 + r * 16 + j + 8 * g;
            size_t row = (size_t)b * SEQ + n;
#pragma unroll
            for (int db = 0; db < 4; ++db)
                abuf[row * CDIM + h * HEAD_DIM + db * 16 + lr] =
                    f32_to_bf16(o[r][db][j] * rinv);
        }
    }
}

// --------------------------------------------------------- output GEMM ----
// WG = 4 waves; wave tile = 32(M) x 64(N)
__global__ void out_gemm(const unsigned short* __restrict__ abf,
                         const unsigned short* __restrict__ wT,
                         const float* __restrict__ bias,
                         float* __restrict__ out) {
    const int wv = threadIdx.x >> 5;
    const int lane = threadIdx.x & 31;
    const int g = lane >> 4;
    const int lr = lane & 15;
    const int m0 = blockIdx.x * 128 + wv * 32;
    const int n0 = blockIdx.y * 64;

    v8f acc[2][4] = {};
    const unsigned short* arow0 = abf + (size_t)(m0 + lr) * CDIM;
    const unsigned short* arow1 = arow0 + (size_t)16 * CDIM;
    for (int k0 = 0; k0 < CDIM; k0 += 32) {
        __builtin_prefetch(arow0 + k0 + 128, 0, 1);
        __builtin_prefetch(arow1 + k0 + 128, 0, 1);
        FragAB a0, a1;
        a0.q[0] = *reinterpret_cast<const uint4*>(arow0 + k0 + g * 8);
        a0.q[1] = *reinterpret_cast<const uint4*>(arow0 + k0 + 16 + g * 8);
        a1.q[0] = *reinterpret_cast<const uint4*>(arow1 + k0 + g * 8);
        a1.q[1] = *reinterpret_cast<const uint4*>(arow1 + k0 + 16 + g * 8);
#pragma unroll
        for (int nb = 0; nb < 4; ++nb) {
            FragAB bm;
            const unsigned short* bcol =
                wT + (size_t)(n0 + nb * 16 + lr) * CDIM + k0 + g * 8;
            bm.q[0] = *reinterpret_cast<const uint4*>(bcol);
            bm.q[1] = *reinterpret_cast<const uint4*>(bcol + 16);
            acc[0][nb] = __builtin_amdgcn_wmma_f32_16x16x32_bf16(
                false, a0.bf, false, bm.bf, (short)0, acc[0][nb], false, false);
            acc[1][nb] = __builtin_amdgcn_wmma_f32_16x16x32_bf16(
                false, a1.bf, false, bm.bf, (short)0, acc[1][nb], false, false);
        }
    }
#pragma unroll
    for (int nb = 0; nb < 4; ++nb) {
        int col = n0 + nb * 16 + lr;
        float bv = bias[col];
#pragma unroll
        for (int r = 0; r < 2; ++r) {
#pragma unroll
            for (int j = 0; j < 8; ++j) {
                int m = m0 + r * 16 + j + 8 * g;
                out[(size_t)m * CDIM + col] = acc[r][nb][j] + bv;
            }
        }
    }
}

// -------------------------------------------------------------- launch ----
extern "C" void kernel_launch(void* const* d_in, const int* in_sizes, int n_in,
                              void* d_out, int out_size, void* d_ws, size_t ws_size,
                              hipStream_t stream) {
    const float* x      = (const float*)d_in[0];   // [4,2048,1024]
    const float* w_qkv  = (const float*)d_in[1];   // [1024,3072]
    const float* w_out  = (const float*)d_in[2];   // [1024,1024]
    const float* b_out  = (const float*)d_in[3];   // [1024]
    float* out = (float*)d_out;                    // [4,2048,1024]

    unsigned short* ws = (unsigned short*)d_ws;
    const size_t N_X    = (size_t)ROWS * CDIM;       // 8,388,608
    const size_t N_WQKV = (size_t)3 * CDIM * CDIM;   // 3,145,728
    const size_t N_WOUT = (size_t)CDIM * CDIM;       // 1,048,576

    unsigned short* xbf    = ws;
    unsigned short* wqkvT  = xbf   + N_X;
    unsigned short* woutT  = wqkvT + N_WQKV;
    unsigned short* qbuf   = woutT + N_WOUT;
    unsigned short* kbuf   = qbuf  + N_X;
    unsigned short* vtbuf  = kbuf  + N_X;
    unsigned short* abuf   = vtbuf + N_X;

    // prep: convert / transpose to bf16
    cvt_f32_bf16<<<(int)((N_X + 255) / 256), 256, 0, stream>>>(x, xbf, (int)N_X);
    transpose_cvt<<<(int)((N_WQKV + 255) / 256), 256, 0, stream>>>(
        w_qkv, wqkvT, CDIM, 3 * CDIM);
    transpose_cvt<<<(int)((N_WOUT + 255) / 256), 256, 0, stream>>>(
        w_out, woutT, CDIM, CDIM);

    // qkv projection: M=8192, N=3072, K=1024 ; 128x64 WG tiles
    qkv_gemm<<<dim3(ROWS / 128, (3 * CDIM) / 64), 128, 0, stream>>>(
        xbf, wqkvT, qbuf, kbuf, vtbuf);

    // flash attention: one WG per (b,h,128-query block)
    flash_attn<<<BATCH * HEADS * (SEQ / 128), 128, 0, stream>>>(
        qbuf, kbuf, vtbuf, abuf);

    // output projection: M=8192, N=1024, K=1024
    out_gemm<<<dim3(ROWS / 128, CDIM / 64), 128, 0, stream>>>(
        abuf, woutT, b_out, out);
}
